// Attention_72773925864296
// MI455X (gfx1250) — matrix operand order repaired and checked by reading, not compiled, not run on hardware
//
#include <hip/hip_runtime.h>
#include <hip/hip_bf16.h>
#include <math.h>

// ---------------------------------------------------------------------------
// Relative-position attention for MI455X (gfx1250), wave32 + WMMA + TDM.
//
// Rel-pos bias: reference gathers embed_luv at the constant index Sk-1 for all
// k, so the bias is constant along the softmax axis and cancels identically in
// softmax(). Neither returned tensor depends on it -> skipped.
// ---------------------------------------------------------------------------

#define SEQ   2000
#define BATCH 4
#define CDIM  1024
#define HEADS 16
#define HDIM  64
#define MROWS (SEQ * BATCH)          // 8000 GEMM rows (s,b)
#define PSTR  2052                   // LDS score-strip stride (2048 data + 4 pad)
#define TSTR  36                     // LDS GEMM tile stride (32 data + 4 pad halfs)

typedef __attribute__((ext_vector_type(16))) _Float16 v16h;
typedef __attribute__((ext_vector_type(4)))  _Float16 v4h;
typedef __attribute__((ext_vector_type(8)))  float    v8f;
typedef __attribute__((ext_vector_type(4)))  unsigned int u32x4;
typedef __attribute__((ext_vector_type(8)))  int      i32x8;
typedef __attribute__((ext_vector_type(4)))  int      i32x4;

#if defined(__has_builtin)
#if __has_builtin(__builtin_amdgcn_tensor_load_to_lds) && \
    __has_builtin(__builtin_amdgcn_s_wait_tensorcnt)
#define HAVE_TDM 1
#endif
#endif
#ifndef HAVE_TDM
#define HAVE_TDM 0
#endif

// ---- WMMA fragment helpers (CDNA5 wave32 layouts, ISA 7.12.2) --------------
// A (16x32 f16): lane m=lane&15 holds row M=m; group g=lane>>4 selects octet:
// VGPR i<4 -> K=2i+{0,1}+8g ; i>=4 -> K=16+2(i-4)+{0,1}+8g.
__device__ __forceinline__ int a_koff(int i, int g) {
  return (i < 4 ? 2 * i : 16 + 2 * (i - 4)) + 8 * g;
}
// B (32x16 f16): lane n=lane&15 holds col N=n; same K order as A:
// K = a_koff(i, g) + {0,1}.

__device__ __forceinline__ v8f wmma_f16(v16h a, v16h b, v8f c) {
  return __builtin_amdgcn_wmma_f32_16x16x32_f16(false, a, false, b, (short)0, c,
                                                false, false);
}

// ---------------------------------------------------------------------------
// TDM: DMA one 64-row x 32-half tile (global row stride CDIM halfs) into LDS
// with 2 pad DWORDs every 16 DWORDs -> LDS rows at stride TSTR=36 halfs.
// D# packing per ISA 8.3/8.4 (group0: count/lds/global/type, group1: dims).
// This toolchain exposes the 6-arg builtin:
//   (u32x4 group0, i32x8 group1, i32x4 group2, i32x4 group3, i32x8, i32 cpol)
// Groups 2/3 (and the trailing i32x8) are zero for a 2D tile.
// ---------------------------------------------------------------------------
#if HAVE_TDM
__device__ __forceinline__ void tdm_load_tile_f16(unsigned lds_addr,
                                                  const _Float16* gptr) {
  unsigned long long ga = (unsigned long long)(uintptr_t)gptr;
  u32x4 g0 = {};
  g0[0] = 1u;                                         // count = 1 (valid D#)
  g0[1] = lds_addr;                                   // LDS byte address
  g0[2] = (unsigned)(ga & 0xffffffffu);               // global_addr[31:0]
  g0[3] = (unsigned)((ga >> 32) & 0x01ffffffu)        // global_addr[56:32]
        | (2u << 30);                                 // type = 2 (image)
  i32x8 g1 = {};
  g1[0] = (1 << 16)                                   // data_size = 1 (2B)
        | (1 << 20)                                   // pad_enable
        | (3 << 22)                                   // pad_interval: 16 DW
        | (1 << 25);                                  // pad_amount: 2 DW
  const unsigned td0 = 1u << 24, td1 = 1u << 24;      // huge tensor dims (no OOB)
  g1[1] = (int)((td0 & 0xffffu) << 16);               // tensor_dim0[15:0]
  g1[2] = (int)(((td0 >> 16) & 0xffffu) | ((td1 & 0xffffu) << 16));
  g1[3] = (int)(((td1 >> 16) & 0xffffu) | (32u << 16)); // tile_dim0 = 32
  g1[4] = (int)64;                                    // tile_dim1=64, tile_dim2=0
  g1[5] = (int)(unsigned)CDIM;                        // tensor_dim0_stride lo
  g1[6] = 0;
  g1[7] = 0;
  i32x4 z4 = {0, 0, 0, 0};
  i32x8 z8 = {};
  __builtin_amdgcn_tensor_load_to_lds(g0, g1, z4, z4, z8, 0);
}
#endif

// ---------------------------------------------------------------------------
// fp32 -> f16 one-pass converter (4 elements/thread)
// ---------------------------------------------------------------------------
__global__ void cvt_f32_to_f16(const float* __restrict__ s,
                               _Float16* __restrict__ d, int n4) {
  int i = blockIdx.x * blockDim.x + threadIdx.x;
  if (i < n4) {
    float4 v = ((const float4*)s)[i];
    v4h h;
    h[0] = (_Float16)v.x; h[1] = (_Float16)v.y;
    h[2] = (_Float16)v.z; h[3] = (_Float16)v.w;
    ((v4h*)d)[i] = h;
  }
}

// ---------------------------------------------------------------------------
// GEMM: Y[M,N] = A[M,K] * W[N,K]^T, all-f16 inputs, f32 accum.
// A: f16 [M=8000, K=1024] row-major.  W: f16 [N=1024, K=1024] row-major.
//  OUTMODE 0 -> f16 ws [B,H,S,D] (Q / pre-scaled K)
//          1 -> f16 ws [B,H,D,S] (V transposed for PV B-fragments)
//          2 -> f32 row-major [M,N] == [S,B,C] (final projection)
// 128 threads (4 waves); 64x64 tile; k-step 32; double-buffered TDM staging.
// ---------------------------------------------------------------------------
template <int OUTMODE>
__global__ void gemm_wmma_kernel(const _Float16* __restrict__ Ag,
                                 const _Float16* __restrict__ Wh,
                                 void* __restrict__ dst, float scale) {
  __shared__ _Float16 As[2][64][TSTR];
  __shared__ _Float16 Bs[2][64][TSTR];   // Bs[n][k] = W[n][k] (no transpose)

  const int mt = blockIdx.x, nt = blockIdx.y;
  const int tid = threadIdx.x, wave = tid >> 5, lane = tid & 31;
  const int g = lane >> 4, ml = lane & 15;
  const int NK = CDIM / 32;

  const _Float16* Abase = Ag + (size_t)(mt * 64) * CDIM;
  const _Float16* Wbase = Wh + (size_t)(nt * 64) * CDIM;

  v8f acc[4] = {v8f{}, v8f{}, v8f{}, v8f{}};

#if HAVE_TDM
  // wave0 DMAs the A tile, wave1 the W tile; waves track their own TENSORcnt.
  auto issue = [&](int kt) {
    int buf = kt & 1;
    if (wave == 0)
      tdm_load_tile_f16((unsigned)(uintptr_t)&As[buf][0][0], Abase + kt * 32);
    else if (wave == 1)
      tdm_load_tile_f16((unsigned)(uintptr_t)&Bs[buf][0][0], Wbase + kt * 32);
  };
#else
  auto issue = [&](int kt) {   // cooperative fallback staging
    int buf = kt & 1;
    for (int idx = tid; idx < 64 * 32; idx += 128) {
      int r = idx >> 5, k = idx & 31;
      As[buf][r][k] = Abase[(size_t)r * CDIM + kt * 32 + k];
      Bs[buf][r][k] = Wbase[(size_t)r * CDIM + kt * 32 + k];
    }
  };
#endif

  issue(0);
  for (int kt = 0; kt < NK; ++kt) {
    const int buf = kt & 1;
#if HAVE_TDM
    __builtin_amdgcn_s_wait_tensorcnt(0);  // this wave's DMA (if any) landed
#endif
    __syncthreads();                       // tile visible; prev compute done
    if (kt + 1 < NK) issue(kt + 1);        // overlap next DMA with compute

    v16h a;
#pragma unroll
    for (int i = 0; i < 8; ++i) {
      int kb = a_koff(i, g);
      a[2 * i]     = As[buf][wave * 16 + ml][kb];
      a[2 * i + 1] = As[buf][wave * 16 + ml][kb + 1];
    }
#pragma unroll
    for (int c4 = 0; c4 < 4; ++c4) {
      v16h b;
#pragma unroll
      for (int i = 0; i < 8; ++i) {
        int k = a_koff(i, g);
        b[2 * i]     = Bs[buf][c4 * 16 + ml][k];
        b[2 * i + 1] = Bs[buf][c4 * 16 + ml][k + 1];
      }
      acc[c4] = wmma_f16(a, b, acc[c4]);
    }
    __syncthreads();                       // done with buf before it's re-DMAd
  }

  // Store C tiles (VGPR v -> M = v + 8g, lane -> N = ml)
#pragma unroll
  for (int c4 = 0; c4 < 4; ++c4) {
#pragma unroll
    for (int v = 0; v < 8; ++v) {
      int m = mt * 64 + wave * 16 + v + 8 * g;   // row = s*BATCH + b
      int col = nt * 64 + c4 * 16 + ml;          // output channel o
      float val = acc[c4][v] * scale;
      if (OUTMODE == 2) {
        ((float*)dst)[(size_t)m * CDIM + col] = val;
      } else {
        int s = m >> 2, bb = m & 3;              // BATCH = 4
        int h = col >> 6, d = col & 63;          // HDIM = 64
        if (OUTMODE == 0)
          ((_Float16*)dst)[(((size_t)(bb * HEADS + h)) * SEQ + s) * HDIM + d] =
              (_Float16)val;
        else
          ((_Float16*)dst)[(((size_t)(bb * HEADS + h)) * HDIM + d) * SEQ + s] =
              (_Float16)val;
      }
    }
  }
}

// ---------------------------------------------------------------------------
// Fused attention: per block -> one (b,h), one 16-row q-tile.
//   scores(WMMA) -> LDS strip [16][PSTR] -> softmax -> single write pass over
//   the 1 GB attn tensor -> P*V (WMMA) -> f16 ws [M, C] for final projection.
// 128 threads (4 waves). Dynamic LDS: 16*PSTR*4 = 131328 B (<320 KB WGP LDS).
// ---------------------------------------------------------------------------
__global__ void attn_fused_kernel(const _Float16* __restrict__ q_ws,
                                  const _Float16* __restrict__ k_ws,
                                  const _Float16* __restrict__ v_ws,
                                  float* __restrict__ attn_out,
                                  _Float16* __restrict__ ao_ws) {
  extern __shared__ float Pbuf[];  // [16][PSTR]
  const int qt = blockIdx.x, bh = blockIdx.y;
  const int tid = threadIdx.x, wave = tid >> 5, lane = tid & 31;
  const int g = lane >> 4, ml = lane & 15;
  const int q0 = qt * 16;

  const _Float16* qb = q_ws + (size_t)bh * SEQ * HDIM;
  const _Float16* kb = k_ws + (size_t)bh * SEQ * HDIM;
  const _Float16* vb = v_ws + (size_t)bh * HDIM * SEQ;

  // Q A-fragments (shared by all waves): row = q0+ml, d contiguous
  v16h a0, a1;
  {
    const _Float16* qr = qb + (size_t)(q0 + ml) * HDIM;
#pragma unroll
    for (int i = 0; i < 8; ++i) {
      int kboff = a_koff(i, g);
      a0[2 * i] = qr[kboff];      a0[2 * i + 1] = qr[kboff + 1];
      a1[2 * i] = qr[32 + kboff]; a1[2 * i + 1] = qr[32 + kboff + 1];
    }
  }

  // ---- scores: S[16, 2000] = Q K^T (K pre-scaled by 1/8) ------------------
  for (int jt = wave; jt < SEQ / 16; jt += 4) {
    const _Float16* kr = kb + (size_t)(jt * 16 + ml) * HDIM;  // col n=ml, k=d
    v16h b0, b1;
#pragma unroll
    for (int i = 0; i < 8; ++i) {
      int d0 = a_koff(i, g);
      b0[2 * i] = kr[d0];      b0[2 * i + 1] = kr[d0 + 1];
      b1[2 * i] = kr[32 + d0]; b1[2 * i + 1] = kr[32 + d0 + 1];
    }
    v8f c = {};
    c = wmma_f16(a0, b0, c);
    c = wmma_f16(a1, b1, c);
#pragma unroll
    for (int v = 0; v < 8; ++v)
      Pbuf[(v + 8 * g) * PSTR + jt * 16 + ml] = c[v];
  }
  __syncthreads();

  // ---- softmax over 2000 cols: 8 lanes per row, shuffle reductions --------
  {
    const int row = tid >> 3, t8 = tid & 7;
    float* pr = Pbuf + row * PSTR;
    float mx = -3.0e38f;
    for (int c = t8; c < SEQ; c += 8) mx = fmaxf(mx, pr[c]);
#pragma unroll
    for (int m = 1; m <= 4; m <<= 1) mx = fmaxf(mx, __shfl_xor(mx, m));
    float sum = 0.f;
    for (int c = t8; c < SEQ; c += 8) {
      float e = __expf(pr[c] - mx);
      pr[c] = e;
      sum += e;
    }
#pragma unroll
    for (int m = 1; m <= 4; m <<= 1) sum += __shfl_xor(sum, m);
    float inv = 1.0f / sum;
    float* ar = attn_out + ((size_t)bh * SEQ + (q0 + row)) * SEQ;
    for (int c = t8; c < SEQ; c += 8) {
      float p = pr[c] * inv;
      pr[c] = p;
      ar[c] = p;                 // the single pass over the 1 GB attn tensor
    }
    for (int c = SEQ + t8; c < 2048; c += 8) pr[c] = 0.f;  // zero-pad tail
  }
  __syncthreads();

  // ---- out[16,64] = P[16,2048] * V[2048,64]; wave w -> d-slice [16w,16w+16)
  const int n0 = wave * 16;
  v8f c = {};
  const _Float16* vcol = vb + (size_t)(n0 + ml) * SEQ;  // col n = d, k = s
  for (int k0 = 0; k0 < 2048; k0 += 32) {
    v16h pa;
#pragma unroll
    for (int i = 0; i < 8; ++i) {
      int kboff = k0 + a_koff(i, g);
      pa[2 * i]     = (_Float16)Pbuf[ml * PSTR + kboff];
      pa[2 * i + 1] = (_Float16)Pbuf[ml * PSTR + kboff + 1];
    }
    v16h pv;
#pragma unroll
    for (int i = 0; i < 8; ++i) {
      int s = k0 + a_koff(i, g);
      int s0 = (s < SEQ - 1) ? s : (SEQ - 1);        // P==0 past SEQ, clamp V
      int s1 = (s + 1 < SEQ - 1) ? (s + 1) : (SEQ - 1);
      pv[2 * i] = vcol[s0];
      pv[2 * i + 1] = vcol[s1];
    }
    c = wmma_f16(pa, pv, c);
  }
  const int bb = bh >> 4, h = bh & 15;               // bh = b*HEADS + h
#pragma unroll
  for (int v = 0; v < 8; ++v) {
    int qr = q0 + v + 8 * g;
    int o = h * 64 + n0 + ml;
    ao_ws[((size_t)qr * BATCH + bb) * CDIM + o] = (_Float16)c[v];
  }
}

// ---------------------------------------------------------------------------
extern "C" void kernel_launch(void* const* d_in, const int* in_sizes, int n_in,
                              void* d_out, int out_size, void* d_ws,
                              size_t ws_size, hipStream_t stream) {
  (void)in_sizes; (void)n_in; (void)out_size; (void)ws_size;
  const float* query = (const float*)d_in[0];
  const float* key   = (const float*)d_in[1];
  const float* value = (const float*)d_in[2];
  const float* Wq    = (const float*)d_in[3];
  const float* Wk    = (const float*)d_in[4];
  const float* Wv    = (const float*)d_in[5];
  // d_in[6] = embed_luv: constant along softmax axis -> cancels exactly
  const float* Wproj = (const float*)d_in[7];

  const size_t HEADEL = (size_t)BATCH * HEADS * SEQ * HDIM;  // 8,192,000
  const int nX = MROWS * CDIM;   // 8,192,000
  const int nW = CDIM * CDIM;    // 1,048,576

  _Float16* q_ws  = (_Float16*)d_ws;
  _Float16* k_ws  = q_ws + HEADEL;
  _Float16* v_ws  = k_ws + HEADEL;
  _Float16* ao_ws = v_ws + HEADEL;        // [8000,1024] f16
  _Float16* x16   = ao_ws + HEADEL;       // reused per input conversion
  _Float16* w16   = x16 + (size_t)nX;     // reused per weight conversion

  float* out  = (float*)d_out;                        // [S,B,C]
  float* attn = out + (size_t)SEQ * BATCH * CDIM;     // [B,H,S,S]

  dim3 ggrid(MROWS / 64, CDIM / 64);  // 125 x 16
  dim3 cblk(256);

  cvt_f32_to_f16<<<dim3(nX / 4 / 256), cblk, 0, stream>>>(query, x16, nX / 4);
  cvt_f32_to_f16<<<dim3(nW / 4 / 256), cblk, 0, stream>>>(Wq, w16, nW / 4);
  gemm_wmma_kernel<0><<<ggrid, 128, 0, stream>>>(x16, w16, q_ws, 1.0f);

  cvt_f32_to_f16<<<dim3(nX / 4 / 256), cblk, 0, stream>>>(key, x16, nX / 4);
  cvt_f32_to_f16<<<dim3(nW / 4 / 256), cblk, 0, stream>>>(Wk, w16, nW / 4);
  gemm_wmma_kernel<0><<<ggrid, 128, 0, stream>>>(x16, w16, k_ws, 0.125f);

  cvt_f32_to_f16<<<dim3(nX / 4 / 256), cblk, 0, stream>>>(value, x16, nX / 4);
  cvt_f32_to_f16<<<dim3(nW / 4 / 256), cblk, 0, stream>>>(Wv, w16, nW / 4);
  gemm_wmma_kernel<1><<<ggrid, 128, 0, stream>>>(x16, w16, v_ws, 1.0f);

  attn_fused_kernel<<<dim3(SEQ / 16, BATCH * HEADS), 128, 16 * PSTR * 4,
                      stream>>>(q_ws, k_ws, v_ws, attn, ao_ws);

  cvt_f32_to_f16<<<dim3(nW / 4 / 256), cblk, 0, stream>>>(Wproj, w16, nW / 4);
  gemm_wmma_kernel<2><<<ggrid, 128, 0, stream>>>(ao_ws, w16, d_out, 1.0f);
}
